// EvoformerBlock_80109730005205
// MI455X (gfx1250) — matrix operand order repaired and checked by reading, not compiled, not run on hardware
//
#include <hip/hip_runtime.h>
#include <hip/hip_bf16.h>

// ---------------------------------------------------------------------------
// Evoformer block for MI455X (gfx1250, wave32, WMMA).
// All GEMMs run through v_wmma_f32_16x16x32_bf16 (bf16 A/B, f32 accum) with
// 4x4 (64x64) per-wave register blocking (4x2 if N%64!=0): 16 WMMAs per 8 fragment loads.
// ---------------------------------------------------------------------------

typedef __bf16 bf16;
typedef __attribute__((ext_vector_type(16))) __bf16 v16bf;
typedef __attribute__((ext_vector_type(8)))  __bf16 v8bf;
typedef __attribute__((ext_vector_type(8)))  float   v8f;

__device__ __forceinline__ bf16 to_bf16(float f) {
  unsigned u = __builtin_bit_cast(unsigned, f);
  unsigned r = u + 0x7fffu + ((u >> 16) & 1u);   // round-to-nearest-even
  unsigned short h = (unsigned short)(r >> 16);
  return __builtin_bit_cast(bf16, h);
}

// ---------------------------------------------------------------------------
// Generic batched WMMA GEMM:  C[z][m][n] = act(alpha * A[z] @ Bt[z] + bias) + res
//   A  : [M,K] bf16 row-major (lda elements per row), per-batch offset sA1/sA2
//   Bt : [K,N] bf16 K-major   (ldb elements per K row), per-batch offset sB1/sB2
//   C  : element (m,n) at  coff + m*ldc + n*ldcn   (f32 and/or bf16 outputs)
//   batch index z = z1*d2 + z2  (grid.z), offsets = z1*s?1 + z2*s?2
// Each wave owns a (BM*16) x (BN*16) macro-tile: BM*BN accumulators reuse
// BM A-fragments and BN B-fragments per K-step (register blocking).
// Fragment layouts per CDNA5 ISA 7.12.2:
//   A (16x32 bf16): lane = 16*(khalf) + m ; elems 0..7 -> K=kh*8.., 8..15 -> +16
//   B (32x16 bf16): lane = 16*(khalf) + n ; elems 0..7 -> K=kh*8.., 8..15 -> +16
//   C (16x16 f32) : vgpr r -> M = r + 8*(lane>>4), N = lane&15
// ---------------------------------------------------------------------------
template <int BM, int BN>
__global__ void __launch_bounds__(256)
k_gemm(const bf16* __restrict__ A, long lda,
       const bf16* __restrict__ Bt, long ldb,
       const float* __restrict__ bias,
       const float* __restrict__ res,
       float* __restrict__ C32, bf16* __restrict__ C16,
       long ldc, long ldcn,
       int M, int N, int K,
       int act, float alpha,
       int d2, long sA1, long sA2, long sB1, long sB2, long sC1, long sC2)
{
  const int lane = threadIdx.x & 31;
  const int wave = threadIdx.x >> 5;
  const int z    = blockIdx.z;
  const int z1   = z / d2, z2 = z - z1 * d2;

  const bf16* Ab  = A  + (long)z1 * sA1 + (long)z2 * sA2;
  const bf16* Bb  = Bt + (long)z1 * sB1 + (long)z2 * sB2;
  const long coff = (long)z1 * sC1 + (long)z2 * sC2;

  const int ntn   = N / (16 * BN);
  const int tiles = (M / (16 * BM)) * ntn;
  const int tile  = blockIdx.x * 8 + wave;          // wave-uniform
  if (tile >= tiles) return;
  const int tm = tile / ntn, tn = tile - tm * ntn;
  const int m0 = tm * (16 * BM), n0 = tn * (16 * BN);

  const int lr = lane & 15;                          // A row / C col
  const int hi = lane >> 4;                          // K-half select
  const bf16* arow = Ab + (long)(m0 + lr) * lda + hi * 8;
  const bf16* brow = Bb + (long)(hi * 8) * ldb + n0 + lr;  // N col = lane&15, K = hi*8 (+16)

  v8f acc[BM][BN];
#pragma unroll
  for (int i = 0; i < BM; ++i)
#pragma unroll
    for (int j = 0; j < BN; ++j) acc[i][j] = v8f{0.f,0.f,0.f,0.f,0.f,0.f,0.f,0.f};

  for (int kk = 0; kk < K; kk += 32) {
    if (kk + 32 < K) __builtin_prefetch(brow + (long)(kk + 32) * ldb, 0, 0);
    v16bf a[BM], b[BN];
#pragma unroll
    for (int i = 0; i < BM; ++i) {
      const bf16* ap = arow + (long)(i * 16) * lda + kk;
      v8bf a0 = *(const v8bf*)(ap);                  // K = kk+hi*8 .. +7
      v8bf a1 = *(const v8bf*)(ap + 16);             // K = kk+hi*8+16 .. +7
      a[i] = __builtin_shufflevector(a0, a1, 0,1,2,3,4,5,6,7,8,9,10,11,12,13,14,15);
    }
#pragma unroll
    for (int j = 0; j < BN; ++j) {
      const bf16* bp = brow + (long)kk * ldb + j * 16;
      v16bf t;
#pragma unroll
      for (int s = 0; s < 8; ++s) {
        t[s]     = bp[(long)s * ldb];
        t[s + 8] = bp[(long)(16 + s) * ldb];
      }
      b[j] = t;
    }
#pragma unroll
    for (int i = 0; i < BM; ++i)
#pragma unroll
      for (int j = 0; j < BN; ++j)
        acc[i][j] = __builtin_amdgcn_wmma_f32_16x16x32_bf16(
            false, a[i], false, b[j], (short)0, acc[i][j], false, false);
  }

#pragma unroll
  for (int i = 0; i < BM; ++i) {
#pragma unroll
    for (int j = 0; j < BN; ++j) {
      const int n = n0 + j * 16 + lr;
      const float bn = bias ? bias[n] : 0.f;
#pragma unroll
      for (int r = 0; r < 8; ++r) {
        const int m = m0 + i * 16 + r + hi * 8;
        float v = acc[i][j][r] * alpha + bn;
        if (act == 1)      v = v > 0.f ? v : 0.f;            // relu
        else if (act == 2) v = 1.f / (1.f + __expf(-v));     // sigmoid
        const long ci = coff + (long)m * ldc + (long)n * ldcn;
        if (res) v += res[ci];
        if (C32) C32[ci] = v;
        if (C16) C16[ci] = to_bf16(v);
      }
    }
  }
}

// LayerNorm over last dim D (block = D threads, one row per block).
__global__ void k_layernorm(const float* __restrict__ x,
                            const float* __restrict__ g,
                            const float* __restrict__ bta,
                            float* __restrict__ y32, bf16* __restrict__ y16, int D)
{
  const long row = blockIdx.x;
  const int tid = threadIdx.x;
  const float v = x[row * D + tid];
  float s1 = v, s2 = v * v;
  for (int o = 16; o > 0; o >>= 1) { s1 += __shfl_xor(s1, o, 32); s2 += __shfl_xor(s2, o, 32); }
  __shared__ float l1[8], l2[8];
  const int w = tid >> 5, nw = blockDim.x >> 5;
  if ((tid & 31) == 0) { l1[w] = s1; l2[w] = s2; }
  __syncthreads();
  if (tid == 0) { float a = 0, b = 0; for (int i = 0; i < nw; ++i) { a += l1[i]; b += l2[i]; } l1[0] = a; l2[0] = b; }
  __syncthreads();
  const float inv = 1.f / (float)D;
  const float mean = l1[0] * inv;
  const float var  = l2[0] * inv - mean * mean;
  const float rs   = rsqrtf(var + 1e-5f);
  const float o    = (v - mean) * rs * g[tid] + bta[tid];
  if (y32) y32[row * D + tid] = o;
  if (y16) y16[row * D + tid] = to_bf16(o);
}

// Softmax over a row of ncol (block = ncol threads); writes bf16 probs.
__global__ void k_softmax(const float* __restrict__ S, bf16* __restrict__ P, int ncol)
{
  const long row = blockIdx.x;
  const int tid = threadIdx.x;
  const float v = S[row * ncol + tid];
  float m = v;
  for (int o = 16; o > 0; o >>= 1) m = fmaxf(m, __shfl_xor(m, o, 32));
  __shared__ float lm[8], ls[8];
  const int w = tid >> 5, nw = blockDim.x >> 5;
  if ((tid & 31) == 0) lm[w] = m;
  __syncthreads();
  if (tid == 0) { float a = lm[0]; for (int i = 1; i < nw; ++i) a = fmaxf(a, lm[i]); lm[0] = a; }
  __syncthreads();
  const float e = __expf(v - lm[0]);
  float s = e;
  for (int o = 16; o > 0; o >>= 1) s += __shfl_xor(s, o, 32);
  if ((tid & 31) == 0) ls[w] = s;
  __syncthreads();
  if (tid == 0) { float a = 0; for (int i = 0; i < nw; ++i) a += ls[i]; ls[0] = a; }
  __syncthreads();
  P[row * ncol + tid] = to_bf16(e / ls[0]);
}

// Kt[(nb*H+h)][e][s] = qkv[(nb*Seq+s)][D + h*hd + e]   (K made K-major for WMMA B)
__global__ void k_build_kt(const bf16* __restrict__ qkv, bf16* __restrict__ Kt,
                           int NB, int Seq, int D, int H)
{
  long i = (long)blockIdx.x * blockDim.x + threadIdx.x;
  const int hd = D / H;
  const long total = (long)NB * H * hd * Seq;
  if (i >= total) return;
  int s = (int)(i % Seq); long t = i / Seq;
  int e = (int)(t % hd);  t /= hd;
  int h = (int)(t % H);   int nb = (int)(t / H);
  Kt[i] = qkv[((long)nb * Seq + s) * (3L * D) + D + (long)h * hd + e];
}

// out[(j*Ar+i)*D+d] = in[(i*Bc+j)*D+d]
__global__ void k_transpose_bf16(const bf16* __restrict__ in, bf16* __restrict__ out,
                                 int Ar, int Bc, int D)
{
  long i = (long)blockIdx.x * blockDim.x + threadIdx.x;
  const long total = (long)Ar * Bc * D;
  if (i >= total) return;
  int d = (int)(i % D); long t = i / D;
  int b = (int)(t % Bc); int a = (int)(t / Bc);
  out[((long)b * Ar + a) * D + d] = in[i];
}

// out[d][a][b] = in[(swap ? b*L+a : a*L+b)][d]   (d-major operands for tri einsum)
__global__ void k_tri_permute(const bf16* __restrict__ in, bf16* __restrict__ out,
                              int L, int D, int swap)
{
  long i = (long)blockIdx.x * blockDim.x + threadIdx.x;
  const long total = (long)L * L * D;
  if (i >= total) return;
  int b = (int)(i % L); long t = i / L;
  int a = (int)(t % L); int d = (int)(t / L);
  long src = swap ? ((long)b * L + a) : ((long)a * L + b);
  out[i] = in[src * D + d];
}

// dst = a + (b?b:0) + (c, optionally read through [j,i] transposed layout)
__global__ void k_combine(float* __restrict__ dst, const float* __restrict__ a,
                          const float* __restrict__ b, const float* __restrict__ c,
                          int Ar, int Bc, int D, int ctrans)
{
  long i = (long)blockIdx.x * blockDim.x + threadIdx.x;
  const long total = (long)Ar * Bc * D;
  if (i >= total) return;
  float v = a[i];
  if (b) v += b[i];
  if (c) {
    if (ctrans) {
      int d = (int)(i % D); long t = i / D;
      int j = (int)(t % Bc); int ii = (int)(t / Bc);
      v += c[((long)j * Ar + ii) * D + d];
    } else v += c[i];
  }
  dst[i] = v;
}

// dst = p + zn + t * g     (g already sigmoid()ed in the GEMM epilogue)
__global__ void k_tri_gate(float* __restrict__ dst, const float* __restrict__ p,
                           const float* __restrict__ zn, const float* __restrict__ t,
                           const float* __restrict__ g, long n)
{
  long i = (long)blockIdx.x * blockDim.x + threadIdx.x;
  if (i >= n) return;
  dst[i] = p[i] + zn[i] + t[i] * g[i];
}

// Wt[k*N+n] = bf16(W[n*K+k])   (weights -> K-major bf16, done once per call)
__global__ void k_wt(const float* __restrict__ W, bf16* __restrict__ Wt, int N, int K)
{
  long i = (long)blockIdx.x * blockDim.x + threadIdx.x;
  if (i >= (long)N * K) return;
  int n = (int)(i % N); int k = (int)(i / N);
  Wt[i] = to_bf16(W[(long)n * K + k]);
}

// ---------------------------------------------------------------------------
extern "C" void kernel_launch(void* const* d_in, const int* in_sizes, int n_in,
                              void* d_out, int out_size, void* d_ws, size_t ws_size,
                              hipStream_t stream)
{
  (void)in_sizes; (void)n_in; (void)out_size; (void)ws_size;
  // Problem constants (reference file)
  const int S = 128, L = 256, DM = 256, DP = 128, HM = 8, HT = 4, HD = 32;
  const long MSAE = (long)S * L * DM;      // 8,388,608
  const long PAIRE = (long)L * L * DP;     // 8,388,608
  const float SCL = 0.17677669529663687f;  // 1/sqrt(32)

  auto F = [&](int i) { return (const float*)d_in[i]; };
  float* msaOut  = (float*)d_out;
  float* pairOut = msaOut + MSAE;

  // ---- workspace bump allocator -------------------------------------------
  char* wsp = (char*)d_ws;
  auto alloc = [&](size_t bytes) -> void* {
    void* p = wsp; wsp += (bytes + 255) & ~(size_t)255; return p;
  };
  auto af  = [&](long n) { return (float*)alloc(n * 4); };
  auto ab  = [&](long n) { return (bf16*)alloc(n * 2); };

  // transposed bf16 weights
  bf16 *wMRi = ab(256L*768), *wMRo = ab(256L*256);
  bf16 *wMCi = ab(256L*768), *wMCo = ab(256L*256);
  bf16 *wTOl = ab(128L*128), *wTOr = ab(128L*128), *wTOg = ab(128L*128);
  bf16 *wTIl = ab(128L*128), *wTIr = ab(128L*128), *wTIg = ab(128L*128);
  bf16 *wTRi = ab(128L*384), *wTRo = ab(128L*128);
  bf16 *wTCi = ab(128L*384), *wTCo = ab(128L*128);
  bf16 *wT1  = ab(128L*512), *wT2  = ab(512L*128);

  float* znF  = af(MSAE);          bf16* znB = ab(MSAE);
  bf16*  tB   = ab(MSAE);          bf16* qkv = ab(33554432L);
  bf16*  kt   = ab(MSAE);          float* sc = af(MSAE);
  bf16*  pb   = ab(MSAE);          bf16* oB  = ab(MSAE);
  float* outF = af(MSAE);          float* gateF = af(PAIRE);
  bf16*  leftB = ab(PAIRE);        bf16* rightB = ab(PAIRE);
  bf16*  rtB   = ab(PAIRE);

  auto wtx = [&](const float* W, bf16* Wt, int N, int K) {
    long n = (long)N * K;
    k_wt<<<(unsigned)((n + 255) / 256), 256, 0, stream>>>(W, Wt, N, K);
  };
  wtx(F(4), wMRi, 768, 256);  wtx(F(6), wMRo, 256, 256);
  wtx(F(10), wMCi, 768, 256); wtx(F(12), wMCo, 256, 256);
  wtx(F(16), wTOl, 128, 128); wtx(F(18), wTOr, 128, 128); wtx(F(20), wTOg, 128, 128);
  wtx(F(24), wTIl, 128, 128); wtx(F(26), wTIr, 128, 128); wtx(F(28), wTIg, 128, 128);
  wtx(F(32), wTRi, 384, 128); wtx(F(34), wTRo, 128, 128);
  wtx(F(38), wTCi, 384, 128); wtx(F(40), wTCo, 128, 128);
  wtx(F(44), wT1, 512, 128);  wtx(F(46), wT2, 128, 512);

  // dispatch: 64x64 macro-tile when N%64==0 (all GEMMs except A*V with N=32)
  auto gemm = [&](const bf16* A, long lda, const bf16* Bt, long ldb,
                  const float* bias, const float* res,
                  float* C32, bf16* C16, long ldc, long ldcn,
                  int M, int N, int K, int act, float alpha,
                  int nbatch, int d2,
                  long sA1, long sA2, long sB1, long sB2, long sC1, long sC2) {
    if ((N & 63) == 0) {
      int tiles = (M / 64) * (N / 64);
      dim3 grid((unsigned)((tiles + 7) / 8), 1, (unsigned)nbatch);
      k_gemm<4, 4><<<grid, 256, 0, stream>>>(A, lda, Bt, ldb, bias, res, C32, C16,
                                             ldc, ldcn, M, N, K, act, alpha,
                                             d2, sA1, sA2, sB1, sB2, sC1, sC2);
    } else {
      int tiles = (M / 64) * (N / 32);
      dim3 grid((unsigned)((tiles + 7) / 8), 1, (unsigned)nbatch);
      k_gemm<4, 2><<<grid, 256, 0, stream>>>(A, lda, Bt, ldb, bias, res, C32, C16,
                                             ldc, ldcn, M, N, K, act, alpha,
                                             d2, sA1, sA2, sB1, sB2, sC1, sC2);
    }
  };
  auto ew = [&](long n) { return (unsigned)((n + 255) / 256); };

  // ======================= MSA row attention (Seq=L) =======================
  k_layernorm<<<S * L, DM, 0, stream>>>(F(0), F(2), F(3), znF, znB, DM);
  gemm(znB, DM, wMRi, 3 * DM, F(5), nullptr, nullptr, qkv, 3 * DM, 1,
       S * L, 3 * DM, DM, 0, 1.f, 1, 1, 0, 0, 0, 0, 0, 0);
  k_build_kt<<<ew(MSAE), 256, 0, stream>>>(qkv, kt, S, L, DM, HM);
  for (int nb0 = 0; nb0 < S; nb0 += 16) {
    gemm(qkv + (long)nb0 * L * 3 * DM, 3 * DM, kt + (long)nb0 * HM * HD * L, L,
         nullptr, nullptr, sc, nullptr, L, 1, L, L, HD, 0, SCL, 16 * HM, HM,
         (long)L * 3 * DM, HD, (long)HM * HD * L, (long)HD * L,
         (long)HM * L * L, (long)L * L);
    k_softmax<<<16 * HM * L, L, 0, stream>>>(sc, pb, L);
    gemm(pb, L, qkv + (long)nb0 * L * 3 * DM + 2 * DM, 3 * DM,
         nullptr, nullptr, nullptr, oB + (long)nb0 * L * DM, DM, 1,
         L, HD, L, 0, 1.f, 16 * HM, HM,
         (long)HM * L * L, (long)L * L, (long)L * 3 * DM, HD, (long)L * DM, HD);
  }
  gemm(oB, DM, wMRo, DM, F(7), nullptr, outF, nullptr, DM, 1,
       S * L, DM, DM, 0, 1.f, 1, 1, 0, 0, 0, 0, 0, 0);
  k_combine<<<ew(MSAE), 256, 0, stream>>>(msaOut, znF, nullptr, outF, S, L, DM, 0);

  // ======================= MSA column attention (Seq=S) ====================
  k_layernorm<<<S * L, DM, 0, stream>>>(msaOut, F(8), F(9), znF, znB, DM);
  k_transpose_bf16<<<ew(MSAE), 256, 0, stream>>>(znB, tB, S, L, DM);
  gemm(tB, DM, wMCi, 3 * DM, F(11), nullptr, nullptr, qkv, 3 * DM, 1,
       S * L, 3 * DM, DM, 0, 1.f, 1, 1, 0, 0, 0, 0, 0, 0);
  k_build_kt<<<ew(MSAE), 256, 0, stream>>>(qkv, kt, L, S, DM, HM);
  for (int nb0 = 0; nb0 < L; nb0 += 16) {
    gemm(qkv + (long)nb0 * S * 3 * DM, 3 * DM, kt + (long)nb0 * HM * HD * S, S,
         nullptr, nullptr, sc, nullptr, S, 1, S, S, HD, 0, SCL, 16 * HM, HM,
         (long)S * 3 * DM, HD, (long)HM * HD * S, (long)HD * S,
         (long)HM * S * S, (long)S * S);
    k_softmax<<<16 * HM * S, S, 0, stream>>>(sc, pb, S);
    gemm(pb, S, qkv + (long)nb0 * S * 3 * DM + 2 * DM, 3 * DM,
         nullptr, nullptr, nullptr, oB + (long)nb0 * S * DM, DM, 1,
         S, HD, S, 0, 1.f, 16 * HM, HM,
         (long)HM * S * S, (long)S * S, (long)S * 3 * DM, HD, (long)S * DM, HD);
  }
  gemm(oB, DM, wMCo, DM, F(13), nullptr, outF, nullptr, DM, 1,
       S * L, DM, DM, 0, 1.f, 1, 1, 0, 0, 0, 0, 0, 0);
  k_combine<<<ew(MSAE), 256, 0, stream>>>(msaOut, znF, nullptr, outF, S, L, DM, 1);

  // ======================= Triangle mult (outgoing, ingoing) ===============
  for (int stage = 0; stage < 2; ++stage) {
    const float* pin = stage == 0 ? F(1) : pairOut;
    int gb = stage == 0 ? 14 : 22;
    bf16 *wl = stage == 0 ? wTOl : wTIl, *wr = stage == 0 ? wTOr : wTIr,
         *wg = stage == 0 ? wTOg : wTIg;
    k_layernorm<<<L * L, DP, 0, stream>>>(pin, F(gb), F(gb + 1), znF, znB, DP);
    gemm(znB, DP, wl, DP, F(gb + 3), nullptr, nullptr, leftB, DP, 1,
         L * L, DP, DP, 0, 1.f, 1, 1, 0, 0, 0, 0, 0, 0);
    gemm(znB, DP, wr, DP, F(gb + 5), nullptr, nullptr, rightB, DP, 1,
         L * L, DP, DP, 0, 1.f, 1, 1, 0, 0, 0, 0, 0, 0);
    gemm(znB, DP, wg, DP, F(gb + 7), nullptr, gateF, nullptr, DP, 1,
         L * L, DP, DP, 2, 1.f, 1, 1, 0, 0, 0, 0, 0, 0);
    // outgoing: A_d[i][k]=left[i,k,d], B_d[k][j]=right[j,k,d] (swap)
    // ingoing : A_d[i][k]=left[k,i,d] (swap), B_d[k][j]=right[k,j,d]
    k_tri_permute<<<ew(PAIRE), 256, 0, stream>>>(leftB, tB, L, DP, stage == 0 ? 0 : 1);
    k_tri_permute<<<ew(PAIRE), 256, 0, stream>>>(rightB, rtB, L, DP, stage == 0 ? 1 : 0);
    gemm(tB, L, rtB, L, nullptr, nullptr, outF, nullptr,
         (long)L * DP, DP, L, L, L, 0, 1.f, DP, 1,
         (long)L * L, 0, (long)L * L, 0, 1, 0);
    k_tri_gate<<<ew(PAIRE), 256, 0, stream>>>(pairOut, pin, znF, outF, gateF, PAIRE);
  }

  // ======================= Triangle attention (row, col) ===================
  for (int stage = 0; stage < 2; ++stage) {
    int gb = stage == 0 ? 30 : 36;
    bf16 *wi = stage == 0 ? wTRi : wTCi, *wo = stage == 0 ? wTRo : wTCo;
    k_layernorm<<<L * L, DP, 0, stream>>>(pairOut, F(gb), F(gb + 1), znF, znB, DP);
    const bf16* xin = znB;
    if (stage == 1) {  // column attention works on the transposed pair
      k_transpose_bf16<<<ew(PAIRE), 256, 0, stream>>>(znB, tB, L, L, DP);
      xin = tB;
    }
    gemm(xin, DP, wi, 3 * DP, F(gb + 3), nullptr, nullptr, qkv, 3 * DP, 1,
         L * L, 3 * DP, DP, 0, 1.f, 1, 1, 0, 0, 0, 0, 0, 0);
    k_build_kt<<<ew(PAIRE), 256, 0, stream>>>(qkv, kt, L, L, DP, HT);
    for (int nb0 = 0; nb0 < L; nb0 += 16) {
      gemm(qkv + (long)nb0 * L * 3 * DP, 3 * DP, kt + (long)nb0 * HT * HD * L, L,
           nullptr, nullptr, sc, nullptr, L, 1, L, L, HD, 0, SCL, 16 * HT, HT,
           (long)L * 3 * DP, HD, (long)HT * HD * L, (long)HD * L,
           (long)HT * L * L, (long)L * L);
      k_softmax<<<16 * HT * L, L, 0, stream>>>(sc, pb, L);
      gemm(pb, L, qkv + (long)nb0 * L * 3 * DP + 2 * DP, 3 * DP,
           nullptr, nullptr, nullptr, oB + (long)nb0 * L * DP, DP, 1,
           L, HD, L, 0, 1.f, 16 * HT, HT,
           (long)HT * L * L, (long)L * L, (long)L * 3 * DP, HD, (long)L * DP, HD);
    }
    gemm(oB, DP, wo, DP, F(gb + 5), nullptr, outF, nullptr, DP, 1,
         L * L, DP, DP, 0, 1.f, 1, 1, 0, 0, 0, 0, 0, 0);
    k_combine<<<ew(PAIRE), 256, 0, stream>>>(pairOut, pairOut, znF, outF,
                                             L, L, DP, stage);
  }

  // ======================= Pair transition =================================
  k_layernorm<<<L * L, DP, 0, stream>>>(pairOut, F(42), F(43), nullptr, znB, DP);
  gemm(znB, DP, wT1, 4 * DP, F(45), nullptr, nullptr, qkv, 4 * DP, 1,
       L * L, 4 * DP, DP, 1, 1.f, 1, 1, 0, 0, 0, 0, 0, 0);
  gemm(qkv, 4 * DP, wT2, DP, F(47), pairOut, pairOut, nullptr, DP, 1,
       L * L, DP, 4 * DP, 0, 1.f, 1, 1, 0, 0, 0, 0, 0, 0);
}
